// NonLocalBlock_8409545965719
// MI455X (gfx1250) — matrix operand order repaired and checked by reading, not compiled, not run on hardware
//
#include <hip/hip_runtime.h>
#include <hip/hip_bf16.h>
#include <math.h>

// ---------------------------------------------------------------------------
// NonLocalBlock for MI455X (gfx1250): bf16 WMMA path, f32 accumulate.
//   B=8, L=2048, C=512, CI=256
// GEMM scheme: 64x64 wave tile (4x4 v_wmma_f32_16x16x32_bf16 tiles),
// 4 waves/block (2x2) -> 128x128 block tile. 16 WMMA per 16 b128 loads.
// __launch_bounds__(128, 1) so the allocator can keep all 16 accumulators
// (128 VGPRs) + 8 operand fragments (64 VGPRs) resident -- no spills.
// ---------------------------------------------------------------------------

typedef __attribute__((ext_vector_type(8)))  float  v8f;
typedef __attribute__((ext_vector_type(16))) __bf16 v16bf;
typedef __attribute__((ext_vector_type(8)))  __bf16 v8bf;

union BF16x16 { v16bf v; v8bf h[2]; };

static __device__ __forceinline__ __bf16 f2bf(float f) { return (__bf16)f; }
static __device__ __forceinline__ float  bf2f(__bf16 b) { return (float)b; }

// ---------------- WMMA fragment loads (wave32, ISA 7.12.2 layouts) ----------

// A fragment 16x32 bf16 from row-major [*, lda] (K contiguous).
// Lane L (0-15): M=L, K = 0..7 and 16..23 ; lane L+16: M=L, K = 8..15 and 24..31.
static __device__ __forceinline__ v16bf load_a_frag(const __bf16* A, int lda,
                                                    int k, int lane) {
    const int m  = lane & 15;
    const int kb = (lane & 16) ? 8 : 0;
    const __bf16* p = A + (size_t)m * lda + (k + kb);
    BF16x16 u;
    u.h[0] = *(const v8bf*)(p);        // K = kb .. kb+7
    u.h[1] = *(const v8bf*)(p + 16);   // K = kb+16 .. kb+23
    return u.v;
}

// B fragment 32x16 bf16; operand stored as [N rows x K cols] row-major
// (column n of B is contiguous in memory).
// Lane n (0-15): K=0..7 and 16..23 of column n ; lane n+16: K=8..15 and 24..31.
static __device__ __forceinline__ v16bf load_b_frag(const __bf16* Bm, int ldb,
                                                    int k, int lane) {
    const int n  = lane & 15;
    const int kb = (lane & 16) ? 8 : 0;
    const __bf16* p = Bm + (size_t)n * ldb + (k + kb);
    BF16x16 u;
    u.h[0] = *(const v8bf*)(p);        // K = kb .. kb+7
    u.h[1] = *(const v8bf*)(p + 16);   // K = kb+16 .. kb+23
    return u.v;
}

// 64x64 wave tile: 4x4 WMMA tiles, K-loop in steps of 32.
// A points at tile row origin (m0), Bm at operand row origin (n0).
static __device__ __forceinline__ void wave_gemm_64x64(
    const __bf16* __restrict__ A, int lda,
    const __bf16* __restrict__ Bm, int ldb,
    int K, v8f acc[4][4]) {
    const int lane = threadIdx.x & 31;
    for (int k = 0; k < K; k += 32) {
        v16bf a[4], b[4];
        #pragma unroll
        for (int i = 0; i < 4; ++i)
            a[i] = load_a_frag(A + (size_t)(i * 16) * lda, lda, k, lane);
        #pragma unroll
        for (int j = 0; j < 4; ++j)
            b[j] = load_b_frag(Bm + (size_t)(j * 16) * ldb, ldb, k, lane);
        #pragma unroll
        for (int i = 0; i < 4; ++i)
            #pragma unroll
            for (int j = 0; j < 4; ++j)
                acc[i][j] = __builtin_amdgcn_wmma_f32_16x16x32_bf16(
                    false, a[i], false, b[j], (short)0, acc[i][j], false, false);
    }
}

// ---------------- conversion kernels ---------------------------------------

__global__ __launch_bounds__(256)
void cvt_f32_bf16_kernel(const float* __restrict__ src,
                         __bf16* __restrict__ dst, int n) {
    int i = blockIdx.x * 256 + threadIdx.x;
    if (i < n) dst[i] = f2bf(src[i]);
}

// W [512 x Ncols] row-major  ->  WT [Ncols x 512] row-major (bf16)
__global__ __launch_bounds__(256)
void cvt_wT_kernel(const float* __restrict__ W,
                   __bf16* __restrict__ WT, int Ncols, int total) {
    int i = blockIdx.x * 256 + threadIdx.x;
    if (i < total) {
        int c = i / Ncols, n = i % Ncols;
        WT[(size_t)n * 512 + c] = f2bf(W[i]);
    }
}

// ---------------- GEMM kernels ---------------------------------------------
// Block = 128 threads = 4 waves arranged 2(M) x 2(N); block tile 128x128.

__global__ __launch_bounds__(128, 1)
void gemm_proj_kernel(const __bf16* __restrict__ X,     // [16384 x 512]
                      const __bf16* __restrict__ WT,    // [1024 x 512]
                      const float* __restrict__ bx,
                      const float* __restrict__ by,
                      const float* __restrict__ bo,
                      __bf16* __restrict__ Px,          // [16384 x 256]
                      __bf16* __restrict__ Py,          // [16384 x 256]
                      __bf16* __restrict__ PoT) {       // [8][512 x 2048]
    const int lane = threadIdx.x & 31;
    const int wave = threadIdx.x >> 5;
    const int m0 = blockIdx.x * 128 + (wave & 1) * 64;
    const int n0 = blockIdx.y * 128 + (wave >> 1) * 64;

    v8f acc[4][4] = {};
    wave_gemm_64x64(X + (size_t)m0 * 512, 512, WT + (size_t)n0 * 512, 512,
                    512, acc);

    const int rowOff = (lane & 16) ? 8 : 0;
    const int colL = lane & 15;
    #pragma unroll
    for (int i = 0; i < 4; ++i) {
        #pragma unroll
        for (int j = 0; j < 4; ++j) {
            const int mt = m0 + i * 16 + rowOff;
            const int n  = n0 + j * 16 + colL;
            const float bias = (n < 256) ? bx[n] : (n < 512) ? by[n - 256] : bo[n - 512];
            #pragma unroll
            for (int r = 0; r < 8; ++r) {
                const int row = mt + r;
                const float v = acc[i][j][r] + bias;
                if (n < 256) {
                    Px[(size_t)row * 256 + n] = f2bf(v);
                } else if (n < 512) {
                    Py[(size_t)row * 256 + (n - 256)] = f2bf(v);
                } else {
                    const int b = row >> 11, l = row & 2047;
                    PoT[((size_t)b * 512 + (n - 512)) * 2048 + l] = f2bf(v);
                }
            }
        }
    }
}

// scores[b,l,m] = sum_i Px[b,l,i] * Py[b,m,i]   (NT GEMM, K=256)
__global__ __launch_bounds__(128, 1)
void gemm_scores_kernel(const __bf16* __restrict__ Px,
                        const __bf16* __restrict__ Py,
                        __bf16* __restrict__ S) {
    const int b = blockIdx.z;
    const __bf16* A  = Px + (size_t)b * 2048 * 256;
    const __bf16* Bm = Py + (size_t)b * 2048 * 256;
    __bf16* out = S + (size_t)b * 2048 * 2048;

    const int lane = threadIdx.x & 31;
    const int wave = threadIdx.x >> 5;
    const int m0 = blockIdx.x * 128 + (wave & 1) * 64;
    const int n0 = blockIdx.y * 128 + (wave >> 1) * 64;

    v8f acc[4][4] = {};
    wave_gemm_64x64(A + (size_t)m0 * 256, 256, Bm + (size_t)n0 * 256, 256,
                    256, acc);

    const int rowOff = (lane & 16) ? 8 : 0;
    const int colL = lane & 15;
    #pragma unroll
    for (int i = 0; i < 4; ++i) {
        #pragma unroll
        for (int j = 0; j < 4; ++j) {
            const int mt = m0 + i * 16 + rowOff;
            const int n  = n0 + j * 16 + colL;
            #pragma unroll
            for (int r = 0; r < 8; ++r)
                out[(size_t)(mt + r) * 2048 + n] = f2bf(acc[i][j][r]);
        }
    }
}

// softmax over the BATCH axis: 8 strided values per (l,m) site, in-place bf16.
__global__ __launch_bounds__(256)
void softmax_batch_kernel(__bf16* __restrict__ S) {
    const size_t idx = (size_t)blockIdx.x * 256 + threadIdx.x;   // < 2048*2048
    const size_t LL = (size_t)2048 * 2048;
    float v[8];
    float mx = -INFINITY;
    #pragma unroll
    for (int b = 0; b < 8; ++b) {
        v[b] = bf2f(S[(size_t)b * LL + idx]);
        mx = fmaxf(mx, v[b]);
    }
    float sum = 0.f;
    #pragma unroll
    for (int b = 0; b < 8; ++b) { v[b] = __expf(v[b] - mx); sum += v[b]; }
    const float inv = 1.0f / sum;
    #pragma unroll
    for (int b = 0; b < 8; ++b)
        S[(size_t)b * LL + idx] = f2bf(v[b] * inv);
}

// s_dot[b,l,c] = sum_m attn[b,l,m] * o_proj[b,m,c] ; out = inputs + s_dot
__global__ __launch_bounds__(128, 1)
void gemm_sdot_kernel(const __bf16* __restrict__ S,    // attn bf16
                      const __bf16* __restrict__ PoT,  // [8][512 x 2048]
                      const float* __restrict__ X,     // inputs f32
                      float* __restrict__ out) {
    const int b = blockIdx.z;
    const __bf16* A  = S + (size_t)b * 2048 * 2048;
    const __bf16* Bm = PoT + (size_t)b * 512 * 2048;

    const int lane = threadIdx.x & 31;
    const int wave = threadIdx.x >> 5;
    const int m0 = blockIdx.x * 128 + (wave & 1) * 64;   // l
    const int n0 = blockIdx.y * 128 + (wave >> 1) * 64;  // c

    v8f acc[4][4] = {};
    wave_gemm_64x64(A + (size_t)m0 * 2048, 2048, Bm + (size_t)n0 * 2048, 2048,
                    2048, acc);

    const int rowOff = (lane & 16) ? 8 : 0;
    const int colL = lane & 15;
    #pragma unroll
    for (int i = 0; i < 4; ++i) {
        #pragma unroll
        for (int j = 0; j < 4; ++j) {
            const int mt = m0 + i * 16 + rowOff;
            const int n  = n0 + j * 16 + colL;
            #pragma unroll
            for (int r = 0; r < 8; ++r) {
                const size_t o = ((size_t)b * 2048 + (mt + r)) * 512 + n;
                out[o] = X[o] + acc[i][j][r];
            }
        }
    }
}

// ---------------- launch ----------------------------------------------------

extern "C" void kernel_launch(void* const* d_in, const int* in_sizes, int n_in,
                              void* d_out, int out_size, void* d_ws, size_t ws_size,
                              hipStream_t stream) {
    const float* inputs = (const float*)d_in[0];   // [8,2048,512]
    const float* Wx = (const float*)d_in[1];       // [512,256]
    const float* bx = (const float*)d_in[2];       // [256]
    const float* Wy = (const float*)d_in[3];       // [512,256]
    const float* by = (const float*)d_in[4];       // [256]
    const float* Wo = (const float*)d_in[5];       // [512,512]
    const float* bo = (const float*)d_in[6];       // [512]
    float* out = (float*)d_out;

    // workspace layout (bytes)
    char* ws = (char*)d_ws;
    __bf16* Xbf  = (__bf16*)(ws);                        // 16,777,216 B
    __bf16* Wcat = (__bf16*)(ws + 16777216);             //  1,048,576 B
    __bf16* Px   = (__bf16*)(ws + 17825792);             //  8,388,608 B
    __bf16* Py   = (__bf16*)(ws + 26214400);             //  8,388,608 B
    __bf16* PoT  = (__bf16*)(ws + 34603008);             // 16,777,216 B
    __bf16* S    = (__bf16*)(ws + 51380224);             // 67,108,864 B
    // total: 118,489,088 bytes

    // 1) inputs -> bf16
    cvt_f32_bf16_kernel<<<32768, 256, 0, stream>>>(inputs, Xbf, 8388608);
    // 2) fused weight operand [1024 x 512], K-contiguous rows
    cvt_wT_kernel<<<512, 256, 0, stream>>>(Wx, Wcat,             256, 131072);
    cvt_wT_kernel<<<512, 256, 0, stream>>>(Wy, Wcat + 256 * 512, 256, 131072);
    cvt_wT_kernel<<<1024, 256, 0, stream>>>(Wo, Wcat + 512 * 512, 512, 262144);
    // 3) fused projection GEMM: [16384x512] @ [512x1024]
    gemm_proj_kernel<<<dim3(128, 8), 128, 0, stream>>>(Xbf, Wcat, bx, by, bo,
                                                       Px, Py, PoT);
    // 4) scores: 8 x [2048x256] @ [256x2048]
    gemm_scores_kernel<<<dim3(16, 16, 8), 128, 0, stream>>>(Px, Py, S);
    // 5) softmax over batch axis
    softmax_batch_kernel<<<16384, 256, 0, stream>>>(S);
    // 6) attn @ o_proj + residual: 8 x [2048x2048] @ [2048x512]
    gemm_sdot_kernel<<<dim3(16, 4, 8), 128, 0, stream>>>(S, PoT, inputs, out);
}
